// Perceiver_21079699488992
// MI455X (gfx1250) — matrix operand order repaired and checked by reading, not compiled, not run on hardware
//
#include <hip/hip_runtime.h>
#include <math.h>

// ---------------------------------------------------------------------------
// Types / WMMA helpers (gfx1250, wave32)
// ---------------------------------------------------------------------------
typedef _Float16 h8   __attribute__((ext_vector_type(8)));
typedef _Float16 v16h __attribute__((ext_vector_type(16)));
typedef float    v8f  __attribute__((ext_vector_type(8)));

__device__ __forceinline__ v8f vzero() {
  v8f r;
#pragma unroll
  for (int i = 0; i < 8; ++i) r[i] = 0.f;
  return r;
}

// A-fragment (16x32 f16, MxK): lane<16 holds row=lane K=k0+0..7 and k0+16..23,
// lane>=16 holds row=lane-15? no: row=lane&15, K=k0+8..15 and k0+24..31.
// => per lane: 8 halves at p (=row*lda + k0 + (lane>>4)*8) and 8 at p+16.
__device__ __forceinline__ v16h ld_af(const _Float16* p) {
  h8 lo = *(const h8*)p;
  h8 hi = *(const h8*)(p + 16);
  v16h r;
#pragma unroll
  for (int i = 0; i < 8; ++i) { r[i] = lo[i]; r[i + 8] = hi[i]; }
  return r;
}

// B-fragment (32x16 f16, KxN): lane holds column n=lane&15, K=k0+(lane>>4)*8+0..7
// and +16..23  => 8 halves at p and 8 at p+16 (same order as A).
__device__ __forceinline__ v16h ld_bf(const _Float16* p) {
  h8 lo = *(const h8*)p;
  h8 hi = *(const h8*)(p + 16);
  v16h r;
#pragma unroll
  for (int i = 0; i < 8; ++i) { r[i] = lo[i]; r[i + 8] = hi[i]; }
  return r;
}

__device__ __forceinline__ v8f wmma32(v16h a, v16h b, v8f c) {
  return __builtin_amdgcn_wmma_f32_16x16x32_f16(false, a, false, b, (short)0, c,
                                                false, false);
}

// ---------------------------------------------------------------------------
// Problem constants
// ---------------------------------------------------------------------------
#define NB   512      // batch
#define LN_  256      // latents
#define PIX  1024     // tokens per image
#define KC   128      // padded channel dim (100 -> 128)
#define KT   160      // padded token dim (131 -> 160)
#define KH   416      // padded MLP hidden (400 -> 416)

// ---------------------------------------------------------------------------
// Token featurizer: rgb + Fourier features, f16, zero-padded to 160 cols
// ---------------------------------------------------------------------------
__global__ void tokens_k(const float* __restrict__ x, const float* __restrict__ fm,
                         _Float16* __restrict__ tok) {
  long long idx = (long long)blockIdx.x * blockDim.x + threadIdx.x;
  if (idx >= (long long)NB * PIX * KT) return;
  int c = (int)(idx % KT);
  long long pt = idx / KT;
  int t = (int)(pt & (PIX - 1));
  int b = (int)(pt >> 10);
  int yy = t >> 5, xx = t & 31;
  float val = 0.f;
  if (c < 3) {
    val = x[((long long)b * 3 + c) * PIX + t];
  } else if (c < 131) {
    int f = c - 3;
    bool is_sin = f >= 64;
    if (is_sin) f -= 64;
    float pr = (xx * (1.f / 31.f)) * fm[f] + (yy * (1.f / 31.f)) * fm[64 + f];
    float a = 6.283185307179586f * pr;
    val = is_sin ? sinf(a) : cosf(a);
  }
  tok[idx] = (_Float16)val;
}

// ---------------------------------------------------------------------------
// Weight packer: f32 [N,K] -> f16 [Npad,Kpad] zero-padded.
// mode 1: K remapped to head-padded layout (kpad = h*32+c <- k = h*25+c).
// ---------------------------------------------------------------------------
__global__ void pack_w_k(const float* __restrict__ w, _Float16* __restrict__ o,
                         int N, int K, int Npad, int Kpad, int mode) {
  int idx = blockIdx.x * 256 + threadIdx.x;
  if (idx >= Npad * Kpad) return;
  int n = idx / Kpad, k = idx % Kpad;
  float v = 0.f;
  if (mode == 0) {
    if (n < N && k < K) v = w[(long long)n * K + k];
  } else {
    int h = k >> 5, c = k & 31;
    if (n < N && h < 4 && c < 25) v = w[(long long)n * K + h * 25 + c];
  }
  o[idx] = (_Float16)v;
}

// ---------------------------------------------------------------------------
// lat init: latents + latent_pos broadcast over batch, pad cols zero (f32)
// ---------------------------------------------------------------------------
__global__ void lat_init_k(const float* __restrict__ lats,
                           const float* __restrict__ lpos, float* __restrict__ lat) {
  long long idx = (long long)blockIdx.x * blockDim.x + threadIdx.x;
  if (idx >= (long long)NB * LN_ * KC) return;
  int c = (int)(idx & (KC - 1));
  int i = (int)((idx >> 7) & (LN_ - 1));
  lat[idx] = (c < 100) ? (lats[i * 100 + c] + lpos[i * 100 + c]) : 0.f;
}

__global__ void f32_to_f16_k(const float* __restrict__ a, _Float16* __restrict__ o,
                             long long n) {
  long long idx = (long long)blockIdx.x * blockDim.x + threadIdx.x;
  if (idx < n) o[idx] = (_Float16)a[idx];
}

// ---------------------------------------------------------------------------
// LayerNorm over first 100 cols of a 128-wide f32 row -> f16 (pads -> 0)
// One 32-lane wave per row, 4 rows per 128-thread block.
// ---------------------------------------------------------------------------
__global__ __launch_bounds__(128) void ln_k(const float* __restrict__ lat,
                                            const float* __restrict__ g,
                                            const float* __restrict__ bb,
                                            _Float16* __restrict__ o) {
  int lane = threadIdx.x & 31;
  long long row = (long long)blockIdx.x * 4 + (threadIdx.x >> 5);
  const float* p = lat + row * KC;
  float x0 = p[lane], x1 = p[lane + 32], x2 = p[lane + 64];
  float x3 = (lane < 4) ? p[lane + 96] : 0.f;
  float s = x0 + x1 + x2 + x3;
  float q = x0 * x0 + x1 * x1 + x2 * x2 + x3 * x3;
  for (int m = 16; m; m >>= 1) { s += __shfl_xor(s, m); q += __shfl_xor(q, m); }
  float mu = s * 0.01f;
  float var = q * 0.01f - mu * mu;
  float rs = rsqrtf(var + 1e-5f);
  _Float16* op = o + row * KC;
  float xs[4] = {x0, x1, x2, x3};
#pragma unroll
  for (int j = 0; j < 4; ++j) {
    int c = lane + 32 * j;
    float v = (c < 100) ? ((xs[j] - mu) * rs * g[c] + bb[c]) : 0.f;
    op[c] = (_Float16)v;
  }
}

// ---------------------------------------------------------------------------
// Generic WMMA GEMM:  D[M,N] = A[M,K] * W[N,K]^T (+bias) (+epilogue)
// Block = 4 waves, each wave owns a 16-row strip and NT 16-col tiles.
// EPI: 0=f16 store, 1=f16 transposed store (V^T), 2=f32 store,
//      3=f32 residual add, 4=QKV head-remap scatter, 5=GELU f16 store
// ---------------------------------------------------------------------------
template <int NT, int EPI>
__global__ __launch_bounds__(128) void gemm_k(
    const _Float16* __restrict__ A, int lda, long long sA,
    const _Float16* __restrict__ W, int ldw, const float* __restrict__ bias,
    float* __restrict__ f32io, long long sF, _Float16* __restrict__ o16, int ldo,
    long long sO, _Float16* __restrict__ vth, long long sVth, int N, int K) {
  int b = blockIdx.z;
  int lane = threadIdx.x & 31;
  int wv = threadIdx.x >> 5;
  int m0 = blockIdx.x * 64 + wv * 16;
  int ntb = blockIdx.y * NT;
  const _Float16* Ab = A + (long long)b * sA;
  v8f acc[NT];
#pragma unroll
  for (int t = 0; t < NT; ++t) acc[t] = vzero();
  int arow = lane & 15;
  int alo = (lane >> 4) * 8;
  int bko = (lane >> 4) * 8;
  for (int k0 = 0; k0 < K; k0 += 32) {
    v16h a = ld_af(Ab + (long long)(m0 + arow) * lda + k0 + alo);
#pragma unroll
    for (int t = 0; t < NT; ++t) {
      v16h bb = ld_bf(W + (long long)((ntb + t) * 16 + arow) * ldw + k0 + bko);
      acc[t] = wmma32(a, bb, acc[t]);
    }
  }
  int rb = m0 + (lane >> 4) * 8;
#pragma unroll
  for (int t = 0; t < NT; ++t) {
    int col = (ntb + t) * 16 + (lane & 15);
    float bs = (bias != nullptr && col < N) ? bias[col] : 0.f;
#pragma unroll
    for (int r = 0; r < 8; ++r) {
      int row = rb + r;
      float v = acc[t][r] + bs;  // padded cols: acc==0 (zero W rows), bs==0
      if constexpr (EPI == 0) {
        o16[(long long)b * sO + (long long)row * ldo + col] = (_Float16)v;
      } else if constexpr (EPI == 1) {
        o16[(long long)b * sO + (long long)col * ldo + row] = (_Float16)v;
      } else if constexpr (EPI == 2) {
        f32io[(long long)b * sF + (long long)row * ldo + col] = v;
      } else if constexpr (EPI == 3) {
        long long ix = (long long)b * sF + (long long)row * ldo + col;
        f32io[ix] = f32io[ix] + v;
      } else if constexpr (EPI == 4) {
        if (col < 100) {
          o16[(long long)b * sO + (long long)row * ldo + (col / 25) * 32 +
              col % 25] = (_Float16)v;
        } else if (col < 200) {
          int n2 = col - 100;
          o16[(long long)b * sO + (long long)row * ldo + 128 + (n2 / 25) * 32 +
              n2 % 25] = (_Float16)v;
        } else if (col < 300) {
          int n2 = col - 200;  // V transposed per head: [h][c][key]
          vth[(long long)b * sVth + (long long)(n2 / 25) * (32 * 256) +
              (long long)(n2 % 25) * 256 + row] = (_Float16)v;
        }
      } else {  // GELU (exact, erf)
        float gl = 0.5f * v * (1.f + erff(v * 0.70710678118654752f));
        o16[(long long)b * sO + (long long)row * ldo + col] = (_Float16)gl;
      }
    }
  }
}

// ---------------------------------------------------------------------------
// Fused cross-attention: per block = one image x 16 queries, 1024 keys.
// S strip in LDS (f16), softmax (max/exp/sum), then P·V^T via WMMA.
// ---------------------------------------------------------------------------
__global__ __launch_bounds__(128) void cross_attn_k(
    const _Float16* __restrict__ q, const _Float16* __restrict__ kk,
    const _Float16* __restrict__ vt, _Float16* __restrict__ out) {
  __shared__ _Float16 S[16][1032];  // +8 halves pad -> conflict-free ds reads
  __shared__ float redm[16][8], reds[16][8], rowinv[16];
  int b = blockIdx.y, qb = blockIdx.x;
  int lane = threadIdx.x & 31, wv = threadIdx.x >> 5;
  int arow = lane & 15, alo = (lane >> 4) * 8, bko = (lane >> 4) * 8;

  const _Float16* qp = q + ((long long)b * LN_ + qb * 16) * KC;
  v16h afr[4];
#pragma unroll
  for (int s = 0; s < 4; ++s)
    afr[s] = ld_af(qp + (long long)arow * KC + s * 32 + alo);

  const _Float16* kb = kk + (long long)b * PIX * KC;
  for (int kt = wv * 16; kt < wv * 16 + 16; ++kt) {
    v8f acc = vzero();
#pragma unroll
    for (int s = 0; s < 4; ++s) {
      v16h bb = ld_bf(kb + (long long)(kt * 16 + (lane & 15)) * KC + s * 32 + bko);
      acc = wmma32(afr[s], bb, acc);
    }
    int col = kt * 16 + (lane & 15);
    int rbs = (lane >> 4) * 8;
#pragma unroll
    for (int r = 0; r < 8; ++r) S[rbs + r][col] = (_Float16)(acc[r] * 0.1f);
  }
  __syncthreads();
  {
    int row = threadIdx.x & 15, seg = threadIdx.x >> 4;  // 16 rows x 8 segs
    int k0 = seg * 128;
    float m = -1e30f;
    for (int j = 0; j < 128; ++j) m = fmaxf(m, (float)S[row][k0 + j]);
    redm[row][seg] = m;
    __syncthreads();
    float mr = redm[row][0];
#pragma unroll
    for (int s2 = 1; s2 < 8; ++s2) mr = fmaxf(mr, redm[row][s2]);
    float sum = 0.f;
    for (int j = 0; j < 128; ++j) {
      float p = __expf((float)S[row][k0 + j] - mr);
      sum += p;
      S[row][k0 + j] = (_Float16)p;
    }
    reds[row][seg] = sum;
    __syncthreads();
    if (seg == 0) {
      float tt = 0.f;
#pragma unroll
      for (int s2 = 0; s2 < 8; ++s2) tt += reds[row][s2];
      rowinv[row] = 1.f / tt;
    }
    __syncthreads();
  }
  const _Float16* vb = vt + (long long)b * KC * PIX;  // V^T: [128][1024]
  int n0a = wv * 16, n0b = (wv + 4) * 16;
  v8f a0 = vzero(), a1 = vzero();
  for (int k0 = 0; k0 < 1024; k0 += 32) {
    v16h af = ld_af(&S[arow][k0 + alo]);
    v16h b0 = ld_bf(vb + (long long)(n0a + (lane & 15)) * PIX + k0 + bko);
    a0 = wmma32(af, b0, a0);
    v16h b1 = ld_bf(vb + (long long)(n0b + (lane & 15)) * PIX + k0 + bko);
    a1 = wmma32(af, b1, a1);
  }
  _Float16* op = out + ((long long)b * LN_ + qb * 16) * KC;
  int rbs = (lane >> 4) * 8;
#pragma unroll
  for (int r = 0; r < 8; ++r) {
    int row = rbs + r;
    op[(long long)row * KC + n0a + (lane & 15)] = (_Float16)(a0[r] * rowinv[row]);
    op[(long long)row * KC + n0b + (lane & 15)] = (_Float16)(a1[r] * rowinv[row]);
  }
}

// ---------------------------------------------------------------------------
// Fused latent MHA: per block = (image, head, 32 queries), 256 keys, hd=25->32
// ---------------------------------------------------------------------------
__global__ __launch_bounds__(128) void mha_k(const _Float16* __restrict__ qkvh,
                                             const _Float16* __restrict__ vth,
                                             _Float16* __restrict__ out) {
  __shared__ _Float16 S[32][264];
  __shared__ float redm[32][4], reds[32][4], rowinv[32];
  int qb = blockIdx.x, h = blockIdx.y, b = blockIdx.z;
  int lane = threadIdx.x & 31, wv = threadIdx.x >> 5;
  int qtile = wv >> 1;
  int arow = lane & 15, alo = (lane >> 4) * 8, bko = (lane >> 4) * 8;
  const _Float16* base = qkvh + (long long)b * LN_ * 256;

  v16h a = ld_af(base + (long long)(qb * 32 + qtile * 16 + arow) * 256 + h * 32 + alo);
  for (int kt = (wv & 1) * 8; kt < (wv & 1) * 8 + 8; ++kt) {
    v16h bb = ld_bf(base + (long long)(kt * 16 + (lane & 15)) * 256 + 128 + h * 32 + bko);
    v8f acc = vzero();
    acc = wmma32(a, bb, acc);
    int col = kt * 16 + (lane & 15);
    int rbs = qtile * 16 + (lane >> 4) * 8;
#pragma unroll
    for (int r = 0; r < 8; ++r) S[rbs + r][col] = (_Float16)(acc[r] * 0.2f);
  }
  __syncthreads();
  {
    int row = threadIdx.x & 31, seg = threadIdx.x >> 5;  // 32 rows x 4 segs
    int k0 = seg * 64;
    float m = -1e30f;
    for (int j = 0; j < 64; ++j) m = fmaxf(m, (float)S[row][k0 + j]);
    redm[row][seg] = m;
    __syncthreads();
    float mr = redm[row][0];
#pragma unroll
    for (int s2 = 1; s2 < 4; ++s2) mr = fmaxf(mr, redm[row][s2]);
    float sum = 0.f;
    for (int j = 0; j < 64; ++j) {
      float p = __expf((float)S[row][k0 + j] - mr);
      sum += p;
      S[row][k0 + j] = (_Float16)p;
    }
    reds[row][seg] = sum;
    __syncthreads();
    if (seg == 0) {
      float tt = 0.f;
#pragma unroll
      for (int s2 = 0; s2 < 4; ++s2) tt += reds[row][s2];
      rowinv[row] = 1.f / tt;
    }
    __syncthreads();
  }
  int nt = wv & 1;
  const _Float16* vb = vth + ((long long)b * 4 + h) * 32 * 256;
  v8f acc = vzero();
  for (int k0 = 0; k0 < 256; k0 += 32) {
    v16h af = ld_af(&S[qtile * 16 + arow][k0 + alo]);
    v16h bb = ld_bf(vb + (long long)(nt * 16 + (lane & 15)) * 256 + k0 + bko);
    acc = wmma32(af, bb, acc);
  }
  int col = h * 32 + nt * 16 + (lane & 15);
#pragma unroll
  for (int r = 0; r < 8; ++r) {
    int rowl = qtile * 16 + (lane >> 4) * 8 + r;
    out[((long long)b * LN_ + qb * 32 + rowl) * KC + col] =
        (_Float16)(acc[r] * rowinv[rowl]);
  }
}

// ---------------------------------------------------------------------------
// Final mean over latents -> [B,100] f32
// ---------------------------------------------------------------------------
__global__ __launch_bounds__(128) void mean_k(const float* __restrict__ lat,
                                              float* __restrict__ out) {
  int b = blockIdx.x, c = threadIdx.x;
  const float* p = lat + (long long)b * LN_ * KC + c;
  float s = 0.f;
  for (int i = 0; i < LN_; ++i) s += p[(long long)i * KC];
  if (c < 100) out[b * 100 + c] = s * (1.f / 256.f);
}

// ---------------------------------------------------------------------------
// Host orchestration
// ---------------------------------------------------------------------------
// Workspace layout (bytes, all 16B-aligned). Packed f16 weights first.
static constexpr size_t WOFF_PROJ = 0;
static constexpr size_t WOFF_Q    = WOFF_PROJ + 128 * 160 * 2;
static constexpr size_t WOFF_K    = WOFF_Q + 128 * 128 * 2;
static constexpr size_t WOFF_V    = WOFF_K + 128 * 128 * 2;
static constexpr size_t WOFF_O    = WOFF_V + 128 * 128 * 2;
static constexpr size_t WOFF_IN0  = WOFF_O + 128 * 128 * 2;
static constexpr size_t WOFF_IN1  = WOFF_IN0 + 304 * 128 * 2;
static constexpr size_t WOFF_OUT0 = WOFF_IN1 + 304 * 128 * 2;
static constexpr size_t WOFF_OUT1 = WOFF_OUT0 + 128 * 128 * 2;
static constexpr size_t WOFF_M10  = WOFF_OUT1 + 128 * 128 * 2;
static constexpr size_t WOFF_M11  = WOFF_M10 + 416 * 128 * 2;
static constexpr size_t WOFF_M20  = WOFF_M11 + 416 * 128 * 2;
static constexpr size_t WOFF_M21  = WOFF_M20 + 128 * 416 * 2;

static constexpr size_t O_KBUF = 1u << 20;
static constexpr size_t O_VT   = O_KBUF + (size_t)NB * PIX * KC * 2;
static constexpr size_t O_LAT  = O_VT + (size_t)NB * KC * PIX * 2;
static constexpr size_t O_VTH  = O_LAT + (size_t)NB * LN_ * KC * 4;
static constexpr size_t O_R    = O_VTH + (size_t)NB * 4 * 32 * 256 * 2;
// setup phase inside region R:
static constexpr size_t O_TOK160 = O_R;
static constexpr size_t O_TOK100 = O_R + (size_t)NB * PIX * KT * 2;
// iteration phase inside region R (aliases the token buffers):
static constexpr size_t O_QIN  = O_R;
static constexpr size_t O_QBUF = O_QIN + (size_t)NB * LN_ * KC * 2;
static constexpr size_t O_ATTN = O_QBUF + (size_t)NB * LN_ * KC * 2;
static constexpr size_t O_H16  = O_ATTN + (size_t)NB * LN_ * KC * 2;
static constexpr size_t O_QKVH = O_H16 + (size_t)NB * LN_ * KC * 2;
static constexpr size_t O_MHA  = O_QKVH + (size_t)NB * LN_ * 256 * 2;
static constexpr size_t O_HID  = O_MHA + (size_t)NB * LN_ * KC * 2;

extern "C" void kernel_launch(void* const* d_in, const int* in_sizes, int n_in,
                              void* d_out, int out_size, void* d_ws, size_t ws_size,
                              hipStream_t stream) {
  const float* x       = (const float*)d_in[0];
  const float* latents = (const float*)d_in[1];
  const float* latpos  = (const float*)d_in[2];
  const float* fm      = (const float*)d_in[3];
  const float* proj_w  = (const float*)d_in[4];
  const float* proj_b  = (const float*)d_in[5];
  const float* q_w = (const float*)d_in[6],  *q_b = (const float*)d_in[7];
  const float* k_w = (const float*)d_in[8],  *k_b = (const float*)d_in[9];
  const float* v_w = (const float*)d_in[10], *v_b = (const float*)d_in[11];
  const float* o_w = (const float*)d_in[12], *o_b = (const float*)d_in[13];
  const float* ln1_g = (const float*)d_in[14], *ln1_b = (const float*)d_in[15];
  const float* in_w  = (const float*)d_in[16], *in_b  = (const float*)d_in[17];
  const float* out_w = (const float*)d_in[18], *out_b = (const float*)d_in[19];
  const float* ln2_g = (const float*)d_in[20], *ln2_b = (const float*)d_in[21];
  const float* w1 = (const float*)d_in[22], *b1 = (const float*)d_in[23];
  const float* w2 = (const float*)d_in[24], *b2 = (const float*)d_in[25];
  (void)in_sizes; (void)n_in; (void)out_size; (void)ws_size;

  char* ws = (char*)d_ws;
  auto F16 = [&](size_t o) { return (_Float16*)(ws + o); };
  auto F32 = [&](size_t o) { return (float*)(ws + o); };

  const size_t WIN[2]  = {WOFF_IN0, WOFF_IN1};
  const size_t WOUT[2] = {WOFF_OUT0, WOFF_OUT1};
  const size_t WM1[2]  = {WOFF_M10, WOFF_M11};
  const size_t WM2[2]  = {WOFF_M20, WOFF_M21};

  // --- pack weights to padded f16 ---
  pack_w_k<<<(128 * 160 + 255) / 256, 256, 0, stream>>>(proj_w, F16(WOFF_PROJ), 100, 131, 128, 160, 0);
  pack_w_k<<<(128 * 128 + 255) / 256, 256, 0, stream>>>(q_w, F16(WOFF_Q), 100, 100, 128, 128, 0);
  pack_w_k<<<(128 * 128 + 255) / 256, 256, 0, stream>>>(k_w, F16(WOFF_K), 100, 100, 128, 128, 0);
  pack_w_k<<<(128 * 128 + 255) / 256, 256, 0, stream>>>(v_w, F16(WOFF_V), 100, 100, 128, 128, 0);
  pack_w_k<<<(128 * 128 + 255) / 256, 256, 0, stream>>>(o_w, F16(WOFF_O), 100, 100, 128, 128, 0);
  for (int d = 0; d < 2; ++d) {
    pack_w_k<<<(304 * 128 + 255) / 256, 256, 0, stream>>>(in_w + (size_t)d * 300 * 100, F16(WIN[d]), 300, 100, 304, 128, 0);
    pack_w_k<<<(128 * 128 + 255) / 256, 256, 0, stream>>>(out_w + (size_t)d * 100 * 100, F16(WOUT[d]), 100, 100, 128, 128, 1);
    pack_w_k<<<(416 * 128 + 255) / 256, 256, 0, stream>>>(w1 + (size_t)d * 400 * 100, F16(WM1[d]), 400, 100, 416, 128, 0);
    pack_w_k<<<(128 * 416 + 255) / 256, 256, 0, stream>>>(w2 + (size_t)d * 100 * 400, F16(WM2[d]), 100, 400, 128, 416, 0);
  }

  // --- tokens, token projection, K and V^T (once; reused by all 8 iters) ---
  tokens_k<<<(unsigned)(((long long)NB * PIX * KT) / 256), 256, 0, stream>>>(x, fm, F16(O_TOK160));
  gemm_k<8, 0><<<dim3(16, 1, NB), 128, 0, stream>>>(
      F16(O_TOK160), KT, (long long)PIX * KT, F16(WOFF_PROJ), KT, proj_b,
      nullptr, 0, F16(O_TOK100), KC, (long long)PIX * KC, nullptr, 0, 100, KT);
  gemm_k<8, 0><<<dim3(16, 1, NB), 128, 0, stream>>>(
      F16(O_TOK100), KC, (long long)PIX * KC, F16(WOFF_K), KC, k_b,
      nullptr, 0, F16(O_KBUF), KC, (long long)PIX * KC, nullptr, 0, 100, KC);
  gemm_k<8, 1><<<dim3(16, 1, NB), 128, 0, stream>>>(
      F16(O_TOK100), KC, (long long)PIX * KC, F16(WOFF_V), KC, v_b,
      nullptr, 0, F16(O_VT), PIX, (long long)KC * PIX, nullptr, 0, 100, KC);

  lat_init_k<<<(unsigned)(((long long)NB * LN_ * KC) / 256), 256, 0, stream>>>(
      latents, latpos, F32(O_LAT));

  // zero head-pad holes once per launch (kept zero across the 8 iterations)
  hipMemsetAsync(F16(O_QKVH), 0, (size_t)NB * LN_ * 256 * 2, stream);
  hipMemsetAsync(F16(O_VTH), 0, (size_t)NB * 4 * 32 * 256 * 2, stream);

  const long long nLat = (long long)NB * LN_ * KC;
  for (int it = 0; it < 8; ++it) {
    f32_to_f16_k<<<(unsigned)(nLat / 256), 256, 0, stream>>>(F32(O_LAT), F16(O_QIN), nLat);
    gemm_k<8, 0><<<dim3(4, 1, NB), 128, 0, stream>>>(
        F16(O_QIN), KC, (long long)LN_ * KC, F16(WOFF_Q), KC, q_b,
        nullptr, 0, F16(O_QBUF), KC, (long long)LN_ * KC, nullptr, 0, 100, KC);
    cross_attn_k<<<dim3(16, NB), 128, 0, stream>>>(F16(O_QBUF), F16(O_KBUF),
                                                   F16(O_VT), F16(O_ATTN));
    gemm_k<8, 2><<<dim3(4, 1, NB), 128, 0, stream>>>(
        F16(O_ATTN), KC, (long long)LN_ * KC, F16(WOFF_O), KC, o_b,
        F32(O_LAT), (long long)LN_ * KC, nullptr, KC, 0, nullptr, 0, 100, KC);

    for (int d = 0; d < 2; ++d) {
      ln_k<<<NB * LN_ / 4, 128, 0, stream>>>(F32(O_LAT), ln1_g + d * 100,
                                             ln1_b + d * 100, F16(O_H16));
      gemm_k<19, 4><<<dim3(4, 1, NB), 128, 0, stream>>>(
          F16(O_H16), KC, (long long)LN_ * KC, F16(WIN[d]), KC, in_b + d * 300,
          nullptr, 0, F16(O_QKVH), 256, (long long)LN_ * 256, F16(O_VTH),
          (long long)4 * 32 * 256, 300, KC);
      mha_k<<<dim3(8, 4, NB), 128, 0, stream>>>(F16(O_QKVH), F16(O_VTH), F16(O_MHA));
      gemm_k<8, 3><<<dim3(4, 1, NB), 128, 0, stream>>>(
          F16(O_MHA), KC, (long long)LN_ * KC, F16(WOUT[d]), KC, out_b + d * 100,
          F32(O_LAT), (long long)LN_ * KC, nullptr, KC, 0, nullptr, 0, 100, KC);

      ln_k<<<NB * LN_ / 4, 128, 0, stream>>>(F32(O_LAT), ln2_g + d * 100,
                                             ln2_b + d * 100, F16(O_H16));
      gemm_k<13, 5><<<dim3(4, 2, NB), 128, 0, stream>>>(
          F16(O_H16), KC, (long long)LN_ * KC, F16(WM1[d]), KC, b1 + d * 400,
          nullptr, 0, F16(O_HID), KH, (long long)LN_ * KH, nullptr, 0, 400, KC);
      gemm_k<8, 3><<<dim3(4, 1, NB), 128, 0, stream>>>(
          F16(O_HID), KH, (long long)LN_ * KH, F16(WM2[d]), KH, b2 + d * 100,
          F32(O_LAT), (long long)LN_ * KC, nullptr, KC, 0, nullptr, 0, 100, KH);
    }
  }
  mean_k<<<NB, 128, 0, stream>>>(F32(O_LAT), (float*)d_out);
}
